// ClassicalSTDP_68994354643155
// MI455X (gfx1250) — matrix operand order repaired and checked by reading, not compiled, not run on hardware
//
#include <hip/hip_runtime.h>

// ---- problem constants (from reference) ----
#define B_DIM   512
#define T_DIM   100
#define NPRE    256
#define NPOST   256
#define KDIM    (B_DIM * T_DIM)   // 51200
#define KCHUNKS (KDIM / 32)       // 1600 chunks of K=32
#define KSPLIT  32
#define KC_PER  (KCHUNKS / KSPLIT) // 50

#define A_PLUS   0.005f
#define A_MINUS  0.00525f
// exp(-1/20) (tau_plus == tau_minus == 20, dt == 1)
#define DECAY_P  0.951229424500714f
#define DECAY_M  0.951229424500714f

typedef __bf16 v16bf __attribute__((ext_vector_type(16)));
typedef float  v8f   __attribute__((ext_vector_type(8)));

__device__ __forceinline__ unsigned short f2bf_rne(float f) {
    union { float f; unsigned u; } v; v.f = f;
    unsigned r = v.u + 0x7FFFu + ((v.u >> 16) & 1u);  // round-to-nearest-even
    return (unsigned short)(r >> 16);
}

// ---------------------------------------------------------------------------
// Phase 1a: exponential pre-synaptic trace scan, packed straight into the
// WMMA A-fragment layout (16-bit A 16x32 per CDNA5 ISA):
//   block(mt, kc) = 1024 B;  lane = m + 16*((k&31)>>3 & 1);
//   half = (k&7) + 8*((k>>4)&1);  addr = lane*32 + half*2
// One thread per (b, p); scans T sequentially (the recurrence is serial in t).
// ---------------------------------------------------------------------------
__global__ __launch_bounds__(256) void stdp_trace_pack(
        const float* __restrict__ pre, unsigned short* __restrict__ effA) {
    const int p = threadIdx.x;        // 0..255 pre-neuron
    const int b = blockIdx.x;         // 0..511 batch
    const int mt = p >> 4;
    const int m  = p & 15;
    const float* src = pre + (size_t)b * T_DIM * NPRE + p;
    float cp = 0.0f, cm = 0.0f;       // trace carries (LTP / LTD)
    #pragma unroll 4
    for (int t = 0; t < T_DIM; ++t) {
        float eff = A_PLUS * cp - A_MINUS * cm;   // trace BEFORE update == trace[t]
        float x = src[(size_t)t * NPRE];
        cp = DECAY_P * (cp + x);
        cm = DECAY_M * (cm + x);
        int k  = b * T_DIM + t;
        int kc = k >> 5;
        int kl = k & 31;
        int lane = m + (((kl >> 3) & 1) << 4);
        int half = (kl & 7) + (((kl >> 4) & 1) << 3);
        size_t off = (((size_t)mt * KCHUNKS + kc) << 9) + (size_t)lane * 16 + half;
        effA[off] = f2bf_rne(eff);
    }
}

// ---------------------------------------------------------------------------
// Phase 1b: pack post spikes into WMMA B-fragment layout (16-bit B 32x16):
//   lane = n + 16*((k>>3)&1);  half = (k&7) + 8*((k>>4)&1)
// ---------------------------------------------------------------------------
__global__ __launch_bounds__(256) void stdp_pack_post(
        const float* __restrict__ post, unsigned short* __restrict__ postB) {
    const int q  = threadIdx.x;       // 0..255 post-neuron
    const int k0 = blockIdx.x * 4;    // 4 consecutive k per block
    const int nt = q >> 4;
    const int n  = q & 15;
    #pragma unroll
    for (int i = 0; i < 4; ++i) {
        int k = k0 + i;
        float x = post[(size_t)k * NPOST + q];
        int kc = k >> 5;
        int kl = k & 31;
        int lane = n + (((kl >> 3) & 1) << 4);
        int half = (kl & 7) + (((kl >> 4) & 1) << 3);
        size_t off = (((size_t)nt * KCHUNKS + kc) << 9) + (size_t)lane * 16 + half;
        postB[off] = f2bf_rne(x);
    }
}

// ---------------------------------------------------------------------------
// Phase 2: split-K WMMA GEMM.  512 blocks = 16 output blocks (64x64) x 32 K-
// splits.  8 waves/block; wave w owns M-tile (w&3) and N-tiles {2*(w>>2),
// 2*(w>>2)+1}.  Each K-chunk: one 32B A-fragment load + two 32B B-fragment
// loads per lane, two v_wmma_f32_16x16x32_bf16.  Partials -> workspace.
// ---------------------------------------------------------------------------
__global__ __launch_bounds__(256) void stdp_wmma_gemm(
        const unsigned short* __restrict__ effA,
        const unsigned short* __restrict__ postB,
        float* __restrict__ part) {
    const int wave = threadIdx.x >> 5;
    const int lane = threadIdx.x & 31;
    const int kb = blockIdx.x & (KSPLIT - 1);
    const int ob = blockIdx.x >> 5;            // 0..15 output block
    const int bm = ob & 3;
    const int bn = ob >> 2;
    const int mt  = bm * 4 + (wave & 3);       // global M tile (16 rows)
    const int nt0 = bn * 4 + ((wave >> 2) << 1);

    const unsigned short* ap  = effA  + (((size_t)mt        * KCHUNKS + (size_t)kb * KC_PER) << 9) + (size_t)lane * 16;
    const unsigned short* bp0 = postB + (((size_t)nt0       * KCHUNKS + (size_t)kb * KC_PER) << 9) + (size_t)lane * 16;
    const unsigned short* bp1 = postB + (((size_t)(nt0 + 1) * KCHUNKS + (size_t)kb * KC_PER) << 9) + (size_t)lane * 16;

    v8f c0 = {0.f,0.f,0.f,0.f,0.f,0.f,0.f,0.f};
    v8f c1 = {0.f,0.f,0.f,0.f,0.f,0.f,0.f,0.f};
    for (int kc = 0; kc < KC_PER; ++kc) {
        __builtin_prefetch(ap  + 512 * 8, 0, 0);   // global_prefetch_b8
        __builtin_prefetch(bp0 + 512 * 8, 0, 0);
        __builtin_prefetch(bp1 + 512 * 8, 0, 0);
        v16bf a  = *(const v16bf*)ap;
        v16bf b0 = *(const v16bf*)bp0;
        v16bf b1 = *(const v16bf*)bp1;
        ap += 512; bp0 += 512; bp1 += 512;
        c0 = __builtin_amdgcn_wmma_f32_16x16x32_bf16(false, a, false, b0,
                                                     (short)0, c0, false, false);
        c1 = __builtin_amdgcn_wmma_f32_16x16x32_bf16(false, a, false, b1,
                                                     (short)0, c1, false, false);
    }

    // C/D layout: VGPR r, lanes 0-15 -> M=r, N=lane; lanes 16-31 -> M=r+8.
    float* dst = part + (size_t)kb * (NPRE * NPOST);
    const int col  = lane & 15;
    const int rowb = mt * 16 + ((lane >> 4) << 3);
    #pragma unroll
    for (int r = 0; r < 8; ++r) {
        int row = rowb + r;
        dst[(size_t)row * NPOST + nt0 * 16 + col]       = c0[r];
        dst[(size_t)row * NPOST + (nt0 + 1) * 16 + col] = c1[r];
    }
}

// ---------------------------------------------------------------------------
// Phase 3: deterministic split-K reduction + 1/(B*T) scaling.
// ---------------------------------------------------------------------------
__global__ __launch_bounds__(256) void stdp_reduce(
        const float* __restrict__ part, float* __restrict__ out) {
    const int i = blockIdx.x * 256 + threadIdx.x;   // 0..65535
    float s = 0.0f;
    #pragma unroll
    for (int kb = 0; kb < KSPLIT; ++kb)
        s += part[(size_t)kb * (NPRE * NPOST) + i];
    out[i] = s * (1.0f / (float)KDIM);
}

extern "C" void kernel_launch(void* const* d_in, const int* in_sizes, int n_in,
                              void* d_out, int out_size, void* d_ws, size_t ws_size,
                              hipStream_t stream) {
    (void)in_sizes; (void)n_in; (void)out_size; (void)ws_size;
    const float* pre  = (const float*)d_in[0];   // [B,T,256] f32
    const float* post = (const float*)d_in[1];   // [B,T,256] f32
    // d_in[2] (weights) only defines output shape; update is independent of it.
    float* out = (float*)d_out;                  // [256,256] f32

    // workspace carve-up: A panels (bf16), B panels (bf16), split-K partials (f32)
    unsigned short* effA  = (unsigned short*)d_ws;                      // 26.2 MB
    unsigned short* postB = effA + (size_t)KDIM * NPRE;                 // 26.2 MB
    float* part = (float*)(postB + (size_t)KDIM * NPOST);               //  8.4 MB

    stdp_trace_pack<<<B_DIM, 256, 0, stream>>>(pre, effA);
    stdp_pack_post<<<KDIM / 4, 256, 0, stream>>>(post, postB);
    stdp_wmma_gemm<<<16 * KSPLIT, 256, 0, stream>>>(effA, postB, part);
    stdp_reduce<<<(NPRE * NPOST) / 256, 256, 0, stream>>>(part, out);
}
